// GhostLinear_85392539779593
// MI455X (gfx1250) — matrix operand order repaired and checked by reading, not compiled, not run on hardware
//
#include <hip/hip_runtime.h>

typedef __attribute__((ext_vector_type(16))) _Float16 v16h;
typedef __attribute__((ext_vector_type(8)))  _Float16 v8h;
typedef __attribute__((ext_vector_type(8)))  float    v8f;
typedef __attribute__((ext_vector_type(4)))  float    v4f;
typedef __attribute__((ext_vector_type(4)))  int      v4i;

// Problem dims (fixed by reference)
#define M_DIM 8192   // B*S = 4*2048
#define K_DIM 2048   // IN
#define N_DIM 8192   // OUT

// GEMM tiling
#define BM   128
#define BN   256
#define BK   32
#define LDSS 40                      // padded LDS row stride in halves
#define STAGE_H ((BM + BN) * LDSS)   // halves per pipeline stage (A then B)
#define KSTEPS (K_DIM / BK)

// ---------------------------------------------------------------------------
// CDNA5 async global->LDS copies (ASYNCcnt-tracked), ISA 08_async_tensor §4.
// ---------------------------------------------------------------------------
__device__ __forceinline__ void async_copy_b128(unsigned lds_addr, const _Float16* gptr) {
    asm volatile("global_load_async_to_lds_b128 %0, %1, off"
                 :: "v"(lds_addr), "v"((unsigned long long)(uintptr_t)gptr)
                 : "memory");
}
// Wait until only the most-recent stage (6 copies) is still outstanding.
__device__ __forceinline__ void wait_async6() {
    asm volatile("s_wait_asynccnt 0x6" ::: "memory");
}
__device__ __forceinline__ void wait_async0() {
    asm volatile("s_wait_asynccnt 0x0" ::: "memory");
}

// ---------------------------------------------------------------------------
// Prep 1: dequantize W: wh[o,i] = f16( lut[idx[o,i]] * scale )
// ---------------------------------------------------------------------------
__global__ __launch_bounds__(256)
void ghost_dequant_w(const int* __restrict__ gi, const float* __restrict__ lut,
                     const float* __restrict__ scale, _Float16* __restrict__ wh) {
    __shared__ float slut[16];
    if (threadIdx.x < 15) slut[threadIdx.x] = lut[threadIdx.x] * scale[0];
    __syncthreads();
    const int i = (blockIdx.x * 256 + threadIdx.x) * 8;
    v4i a = __builtin_nontemporal_load((const v4i*)(gi + i));
    v4i b = __builtin_nontemporal_load((const v4i*)(gi + i + 4));
    v8h o;
    o[0] = (_Float16)slut[a.x]; o[1] = (_Float16)slut[a.y];
    o[2] = (_Float16)slut[a.z]; o[3] = (_Float16)slut[a.w];
    o[4] = (_Float16)slut[b.x]; o[5] = (_Float16)slut[b.y];
    o[6] = (_Float16)slut[b.z]; o[7] = (_Float16)slut[b.w];
    *(v8h*)(wh + i) = o;   // regular store: stays hot in L2 for the GEMM
}

// ---------------------------------------------------------------------------
// Prep 2: convert x (f32) -> f16
// ---------------------------------------------------------------------------
__global__ __launch_bounds__(256)
void ghost_cvt_x(const float* __restrict__ x, _Float16* __restrict__ xh) {
    const int i = (blockIdx.x * 256 + threadIdx.x) * 8;
    v4f a = __builtin_nontemporal_load((const v4f*)(x + i));
    v4f b = __builtin_nontemporal_load((const v4f*)(x + i + 4));
    v8h o;
    o[0] = (_Float16)a.x; o[1] = (_Float16)a.y;
    o[2] = (_Float16)a.z; o[3] = (_Float16)a.w;
    o[4] = (_Float16)b.x; o[5] = (_Float16)b.y;
    o[6] = (_Float16)b.z; o[7] = (_Float16)b.w;
    *(v8h*)(xh + i) = o;
}

// ---------------------------------------------------------------------------
// GEMM: C[M,N] = Xh[M,K] * Wh[N,K]^T
// 256 threads = 8 waves. WG tile 128x256, wave tile 64x64 (4x4 WMMA tiles).
// 3-stage LDS pipeline fed by async-to-LDS; partial ASYNCcnt waits give each
// stage two compute iterations of latency slack. One barrier per K-step.
// ---------------------------------------------------------------------------
__global__ __launch_bounds__(256)
void ghost_gemm(const _Float16* __restrict__ A,   // Xh [M,K]
                const _Float16* __restrict__ Bw,  // Wh [N,K]
                float* __restrict__ C) {
    __shared__ _Float16 lds[3 * STAGE_H];   // 3 x 30 KB

    const int tid   = threadIdx.x;
    const int lane  = tid & 31;
    const int wave  = tid >> 5;
    const int waveM = wave >> 2;   // 0..1 -> 64-row half
    const int waveN = wave & 3;    // 0..3 -> 64-col slice

    const int m0 = blockIdx.y * BM;
    const int n0 = blockIdx.x * BN;

    // Staging geometry: chunk = v8h (16B). A: 512 chunks, B: 1024 chunks.
    const int r0 = tid >> 2;           // 0..63
    const int c0 = (tid & 3) * 8;      // 0,8,16,24

    // Per-thread global base pointers (k advances by BK per stage)
    const _Float16* gA0 = A  + (size_t)(m0 + r0)      * K_DIM + c0;
    const _Float16* gA1 = A  + (size_t)(m0 + r0 + 64) * K_DIM + c0;
    const _Float16* gB0 = Bw + (size_t)(n0 + r0)       * K_DIM + c0;
    const _Float16* gB1 = Bw + (size_t)(n0 + r0 + 64)  * K_DIM + c0;
    const _Float16* gB2 = Bw + (size_t)(n0 + r0 + 128) * K_DIM + c0;
    const _Float16* gB3 = Bw + (size_t)(n0 + r0 + 192) * K_DIM + c0;

    // Per-thread stage-relative LDS byte offsets (A block, then B block)
    const unsigned ldsBase = (unsigned)(uintptr_t)(&lds[0]);
    const unsigned uA0 = (unsigned)(((r0)            * LDSS + c0) * 2);
    const unsigned uA1 = (unsigned)(((r0 + 64)       * LDSS + c0) * 2);
    const unsigned uB0 = (unsigned)((BM * LDSS + (r0)       * LDSS + c0) * 2);
    const unsigned uB1 = (unsigned)((BM * LDSS + (r0 + 64)  * LDSS + c0) * 2);
    const unsigned uB2 = (unsigned)((BM * LDSS + (r0 + 128) * LDSS + c0) * 2);
    const unsigned uB3 = (unsigned)((BM * LDSS + (r0 + 192) * LDSS + c0) * 2);
#define STAGE_BYTES ((unsigned)(STAGE_H * 2))

    const int lrow  = lane & 15;       // row/col within 16x16 frag
    const int lhalf = lane >> 4;       // K-half selector per ISA layout

    // Issue one stage's 6 async copies for k-step `kt` into buffer `buf`.
    auto issue_stage = [&](int kt, int buf) {
        const int nk = kt * BK;
        const unsigned sb = ldsBase + (unsigned)buf * STAGE_BYTES;
        async_copy_b128(sb + uA0, gA0 + nk);
        async_copy_b128(sb + uA1, gA1 + nk);
        async_copy_b128(sb + uB0, gB0 + nk);
        async_copy_b128(sb + uB1, gB1 + nk);
        async_copy_b128(sb + uB2, gB2 + nk);
        async_copy_b128(sb + uB3, gB3 + nk);
    };

    v8f acc[4][4] = {};

    // Prologue: two stages in flight.
    issue_stage(0, 0);
    issue_stage(1, 1);

    int cur = 0;    // buffer holding stage kt
    int nxt2 = 2;   // buffer receiving stage kt+2

    for (int kt = 0; kt < KSTEPS; ++kt) {
        // Oldest in-flight stage (kt) must have landed; leave the newest
        // stage's 6 copies outstanding. In-order ASYNCcnt guarantees this.
        if (kt + 1 < KSTEPS) wait_async6(); else wait_async0();
        __syncthreads();   // all waves' copies for stage kt visible; all
                           // reads of stage kt-1 retired -> its buffer free.

        if (kt + 2 < KSTEPS) issue_stage(kt + 2, nxt2);

        const _Float16* sA = lds + cur * STAGE_H;
        const _Float16* sB = sA + BM * LDSS;

        // A fragments (16-bit A 16x32 layout): VGPRs 0-3: K = 8*lhalf..+7,
        // VGPRs 4-7: K = 16+8*lhalf..+7, row = lane%16.
        v16h afrag[4];
#pragma unroll
        for (int mt = 0; mt < 4; ++mt) {
            const _Float16* p = sA + (waveM * 64 + mt * 16 + lrow) * LDSS;
            v8h lo = *(const v8h*)(p + 8 * lhalf);
            v8h hi = *(const v8h*)(p + 16 + 8 * lhalf);
            afrag[mt] = __builtin_shufflevector(lo, hi,
                0,1,2,3,4,5,6,7,8,9,10,11,12,13,14,15);
        }
        // B fragments: col = lane%16; same K layout as A: K = 8*lhalf..+7, 16+8*lhalf..+7.
        v16h bfrag[4];
#pragma unroll
        for (int nt = 0; nt < 4; ++nt) {
            const _Float16* p = sB + (waveN * 64 + nt * 16 + lrow) * LDSS;
            v8h lo = *(const v8h*)(p + 8 * lhalf);
            v8h hi = *(const v8h*)(p + 16 + 8 * lhalf);
            bfrag[nt] = __builtin_shufflevector(lo, hi,
                0,1,2,3,4,5,6,7,8,9,10,11,12,13,14,15);
        }

#pragma unroll
        for (int mt = 0; mt < 4; ++mt)
#pragma unroll
            for (int nt = 0; nt < 4; ++nt)
                acc[mt][nt] = __builtin_amdgcn_wmma_f32_16x16x32_f16(
                    false, afrag[mt], false, bfrag[nt],
                    (short)0, acc[mt][nt], false, false);

        cur  = (cur  == 2) ? 0 : cur  + 1;
        nxt2 = (nxt2 == 2) ? 0 : nxt2 + 1;
    }

    // Epilogue: C/D layout — VGPR r: M = r (lanes 0-15) / r+8 (lanes 16-31),
    // N = lane%16. Streaming 256 MB output -> non-temporal stores keep the
    // f16 operand matrices resident in L2.
#pragma unroll
    for (int mt = 0; mt < 4; ++mt) {
#pragma unroll
        for (int nt = 0; nt < 4; ++nt) {
            const int gm = m0 + waveM * 64 + mt * 16 + lhalf * 8;
            const int gn = n0 + waveN * 64 + nt * 16 + lrow;
#pragma unroll
            for (int r = 0; r < 8; ++r)
                __builtin_nontemporal_store(acc[mt][nt][r],
                    C + (size_t)(gm + r) * N_DIM + gn);
        }
    }
}

// ---------------------------------------------------------------------------
extern "C" void kernel_launch(void* const* d_in, const int* in_sizes, int n_in,
                              void* d_out, int out_size, void* d_ws, size_t ws_size,
                              hipStream_t stream) {
    const float* x     = (const float*)d_in[0];   // [4,2048,2048] f32
    const int*   gidx  = (const int*)  d_in[1];   // [8192,2048]   i32
    const float* scale = (const float*)d_in[2];   // [1]
    const float* lut   = (const float*)d_in[3];   // [15]
    float*       out   = (float*)d_out;           // [4,2048,8192] f32

    _Float16* xh = (_Float16*)d_ws;                       // 32 MB
    _Float16* wh = xh + (size_t)M_DIM * K_DIM;            // 32 MB

    const int cvt_blocks = (M_DIM * K_DIM) / (256 * 8);   // 8192
    ghost_cvt_x<<<cvt_blocks, 256, 0, stream>>>(x, xh);

    const int dq_blocks = (N_DIM * K_DIM) / (256 * 8);    // 8192
    ghost_dequant_w<<<dq_blocks, 256, 0, stream>>>(gidx, lut, scale, wh);

    dim3 grid(N_DIM / BN, M_DIM / BM);                    // 32 x 64
    ghost_gemm<<<grid, 256, 0, stream>>>(xh, wh, out);
}
